// Model_28518582846368
// MI455X (gfx1250) — compile-verified, matrix operand order checked by reading, not run on hardware
//
#include <hip/hip_runtime.h>
#include <hip/hip_bf16.h>

// ---------------------------------------------------------------------------
// MI455X (gfx1250) implementation of the hierarchical-attention network.
// All large matmuls (GRU gates, attention projections) run through
// v_wmma_f32_16x16x32_f16 with fused epilogues. Wave32 layout per CDNA5 ISA.
// ---------------------------------------------------------------------------

typedef __attribute__((ext_vector_type(16))) _Float16 v16h;
typedef __attribute__((ext_vector_type(8)))  _Float16 v8h;
typedef __attribute__((ext_vector_type(8)))  float    v8f;

#define WMMA16(a, b, c) \
  __builtin_amdgcn_wmma_f32_16x16x32_f16(false, (a), false, (b), (short)0, (c), false, false)

// Load one 16x32 f16 fragment (A-layout; B uses the mirrored layout with the
// same loader since weights are stored [out, in] row-major).
// Lane L<16  : row L,   K = kk+0..7   and kk+16..23
// Lane L>=16 : row L-16,K = kk+8..15  and kk+24..31
__device__ __forceinline__ v16h load_frag16(const _Float16* __restrict__ base,
                                            int row_stride, int kk) {
  const int lane = threadIdx.x & 31;
  const int r    = lane & 15;
  const int kh   = (lane >> 4) & 1;
  const _Float16* p = base + (size_t)r * row_stride + kk + kh * 8;
  v8h lo = *(const v8h*)(p);
  v8h hi = *(const v8h*)(p + 16);
  v16h f;
#pragma unroll
  for (int i = 0; i < 8; ++i) { f[i] = lo[i]; f[i + 8] = hi[i]; }
  return f;
}

// ---------------------------------------------------------------------------
// Small utility kernels
// ---------------------------------------------------------------------------
__global__ void k_f32_to_f16(const float* __restrict__ src, _Float16* __restrict__ dst, int n) {
  int i = blockIdx.x * blockDim.x + threadIdx.x;
  if (i < n) dst[i] = (_Float16)src[i];
}

// One block per token (B*S*T = 65536 tokens), 256 threads = embedding dim.
__global__ void k_embed_gather(const int* __restrict__ x, const float* __restrict__ emb,
                               _Float16* __restrict__ xe) {
  const int tok = blockIdx.x;
  const int e   = threadIdx.x;
  const int w   = x[tok];
  xe[(size_t)tok * 256 + e] = (_Float16)emb[(size_t)w * 256 + e];
}

// q_emb[q] = mean over LQ=8 query-token embeddings.
__global__ void k_query_emb(const int* __restrict__ qids, const float* __restrict__ emb,
                            float* __restrict__ qe) {
  const int q = blockIdx.x, e = threadIdx.x;
  float acc = 0.f;
#pragma unroll
  for (int l = 0; l < 8; ++l) acc += emb[(size_t)qids[q * 8 + l] * 256 + e];
  qe[q * 256 + e] = acc * 0.125f;
}

// ---------------------------------------------------------------------------
// Fused GRU step: xg_t = X_t @ Wih^T (K=256) and hg = h @ Whh^T (K=128) via
// WMMA, kept separate (PyTorch n-gate needs xn + r*hn), then gate math.
// One workgroup owns 16 sequences -> hidden state never crosses workgroups.
// grid.x = Nseq/16, block = 256 (8 waves, 3 N-tiles of the 16x384 gates each).
// ---------------------------------------------------------------------------
__global__ void k_gru_step(const _Float16* __restrict__ X, int Tmax, int t_eff,
                           const _Float16* __restrict__ Wih,  // [384,256] f16
                           const _Float16* __restrict__ Whh,  // [384,128] f16
                           const float* __restrict__ bih, const float* __restrict__ bhh,
                           float* __restrict__ H,             // [Nseq,128] f32 state
                           _Float16* __restrict__ H16,        // [Nseq,128] f16 state
                           _Float16* __restrict__ Y,          // [Nseq,Tmax,256] output
                           int ycol) {
  __shared__ float gx[16][384];
  __shared__ float gh[16][384];
  const int wave = threadIdx.x >> 5;
  const int lane = threadIdx.x & 31;
  const int row0 = blockIdx.x * 16;
  const _Float16* xbase = X + ((size_t)row0 * Tmax + t_eff) * 256;
  const _Float16* hbase = H16 + (size_t)row0 * 128;

#pragma unroll
  for (int tt = 0; tt < 3; ++tt) {
    const int nt = wave * 3 + tt;           // gate column tile (24 total)
    v8f cx = {};
    v8f ch = {};
    const _Float16* wb = Wih + (size_t)nt * 16 * 256;
#pragma unroll
    for (int kk = 0; kk < 256; kk += 32) {
      v16h a = load_frag16(xbase, Tmax * 256, kk);
      v16h b = load_frag16(wb, 256, kk);
      cx = WMMA16(a, b, cx);
    }
    const _Float16* wh = Whh + (size_t)nt * 16 * 128;
#pragma unroll
    for (int kk = 0; kk < 128; kk += 32) {
      v16h a = load_frag16(hbase, 128, kk);
      v16h b = load_frag16(wh, 128, kk);
      ch = WMMA16(a, b, ch);
    }
    const int g    = nt * 16 + (lane & 15);
    const int moff = (lane >> 4) * 8;
    const float bi = bih[g], bh = bhh[g];
#pragma unroll
    for (int j = 0; j < 8; ++j) {
      gx[moff + j][g] = cx[j] + bi;
      gh[moff + j][g] = ch[j] + bh;
    }
  }
  __syncthreads();  // all H16 reads done; gates in LDS

#pragma unroll
  for (int rep = 0; rep < 8; ++rep) {
    const int lin = rep * 256 + threadIdx.x;   // 16 rows x 128 hidden
    const int m   = lin >> 7;
    const int hh  = lin & 127;
    const float r = 1.f / (1.f + __expf(-(gx[m][hh]       + gh[m][hh])));
    const float z = 1.f / (1.f + __expf(-(gx[m][128 + hh] + gh[m][128 + hh])));
    const float n = tanhf(gx[m][256 + hh] + r * gh[m][256 + hh]);
    const size_t hidx = (size_t)(row0 + m) * 128 + hh;
    const float hold = H[hidx];
    const float hnew = (1.f - z) * n + z * hold;
    H[hidx]   = hnew;
    H16[hidx] = (_Float16)hnew;
    Y[((size_t)(row0 + m) * Tmax + t_eff) * 256 + ycol + hh] = (_Float16)hnew;
  }
}

// ---------------------------------------------------------------------------
// Fused attention score: out[row] = vec . tanh(W @ f_row + bias) + b2.
// Covers both the unconditional branch (W1, w2) and query branches (Wq, q_emb)
// without ever materializing hq ([5,B,S,T,256]) in memory.
// grid.x = Mrows/16, block = 32 (one wave). 16 column tiles x 8 K-steps WMMA.
// ---------------------------------------------------------------------------
__global__ void k_attn_score(const _Float16* __restrict__ F,    // [Mrows,256]
                             const _Float16* __restrict__ W16,  // [256,256] out x in
                             const float* __restrict__ bias,    // [256]
                             const float* __restrict__ vec,     // [256]
                             const float* __restrict__ b2,      // nullable [1]
                             float* __restrict__ out) {         // [Mrows]
  const int lane = threadIdx.x & 31;
  const int row0 = blockIdx.x * 16;
  const _Float16* fbase = F + (size_t)row0 * 256;

  v16h afr[8];
#pragma unroll
  for (int k = 0; k < 8; ++k) afr[k] = load_frag16(fbase, 256, k * 32);

  float part[8];
#pragma unroll
  for (int j = 0; j < 8; ++j) part[j] = 0.f;

  for (int nt = 0; nt < 16; ++nt) {
    v8f c = {};
    const _Float16* wb = W16 + (size_t)nt * 16 * 256;
#pragma unroll
    for (int k = 0; k < 8; ++k) {
      v16h b = load_frag16(wb, 256, k * 32);
      c = WMMA16(afr[k], b, c);
    }
    const int e = nt * 16 + (lane & 15);
    const float be = bias[e], ve = vec[e];
#pragma unroll
    for (int j = 0; j < 8; ++j) part[j] += tanhf(c[j] + be) * ve;
  }
  // Rows live in vgpr index j (+8 for lanes>=16); N is striped over 16 lanes.
#pragma unroll
  for (int j = 0; j < 8; ++j) {
#pragma unroll
    for (int off = 1; off < 16; off <<= 1) part[j] += __shfl_xor(part[j], off, 16);
  }
  if ((lane & 15) == 0) {
    const float add = b2 ? b2[0] : 0.f;
    const int mo = (lane >> 4) * 8;
#pragma unroll
    for (int j = 0; j < 8; ++j) out[row0 + mo + j] = part[j] + add;
  }
}

// ---------------------------------------------------------------------------
// Six-branch softmax over the time axis + mean. grid = N, block = T (<=64).
// ---------------------------------------------------------------------------
__global__ void k_softmax_mean(const float* __restrict__ s_t, const float* __restrict__ s_q,
                               float* __restrict__ a_mean, float* __restrict__ a_q_out,
                               int N, int T) {
  __shared__ float buf[64];
  const int n = blockIdx.x, t = threadIdx.x;
  float total = 0.f;
  for (int br = 0; br < 6; ++br) {
    const float* s = (br == 0) ? (s_t + (size_t)n * T)
                               : (s_q + ((size_t)(br - 1) * N + n) * T);
    const float v = s[t];
    buf[t] = v; __syncthreads();
    for (int o = T >> 1; o > 0; o >>= 1) {
      if (t < o) buf[t] = fmaxf(buf[t], buf[t + o]);
      __syncthreads();
    }
    const float m = buf[0]; __syncthreads();
    const float e = __expf(v - m);
    buf[t] = e; __syncthreads();
    for (int o = T >> 1; o > 0; o >>= 1) {
      if (t < o) buf[t] += buf[t + o];
      __syncthreads();
    }
    const float a = e / buf[0]; __syncthreads();
    total += a;
    if (br > 0 && a_q_out) a_q_out[((size_t)(br - 1) * N + n) * T + t] = a;
  }
  a_mean[(size_t)n * T + t] = total * (1.f / 6.f);
}

// pooled[n,e] = sum_t a_mean[n,t] * F[n,t,e].  grid = N, block = 256.
__global__ void k_pool(const float* __restrict__ a_mean, const _Float16* __restrict__ F,
                       float* __restrict__ outf, _Float16* __restrict__ outh, int T) {
  const int n = blockIdx.x, e = threadIdx.x;
  float acc = 0.f;
  for (int t = 0; t < T; ++t)
    acc += a_mean[(size_t)n * T + t] * (float)F[((size_t)n * T + t) * 256 + e];
  outf[(size_t)n * 256 + e] = acc;
  if (outh) outh[(size_t)n * 256 + e] = (_Float16)acc;
}

// Classifier (doc @ Wfc^T + bfc, softmax over C=2) + all_q_value. block = 64.
__global__ void k_head(const float* __restrict__ doc, const float* __restrict__ Wfc,
                       const float* __restrict__ bfc, const float* __restrict__ a_q_s,
                       float* __restrict__ out) {
  __shared__ float logits[32][2];
  const int tid = threadIdx.x;
  if (tid < 64) {
    const int b = tid >> 1, c = tid & 1;
    float acc = bfc[c];
    for (int e = 0; e < 256; ++e) acc += doc[b * 256 + e] * Wfc[c * 256 + e];
    logits[b][c] = acc;
  }
  __syncthreads();
  if (tid < 64) {
    const int b = tid >> 1, c = tid & 1;
    const float m  = fmaxf(logits[b][0], logits[b][1]);
    const float e0 = __expf(logits[b][0] - m), e1 = __expf(logits[b][1] - m);
    out[b * 2 + c] = __expf(logits[b][c] - m) / (e0 + e1);
  }
  if (tid < 32) {
    const int b = tid;
    for (int q = 0; q < 5; ++q) {
      float s = 0.f;
      for (int ss = 0; ss < 32; ++ss) s += a_q_s[((size_t)q * 32 + b) * 32 + ss];
      out[64 + b * 5 + q] = s;
    }
  }
}

// ---------------------------------------------------------------------------
// Host orchestration
// ---------------------------------------------------------------------------
extern "C" void kernel_launch(void* const* d_in, const int* in_sizes, int n_in,
                              void* d_out, int out_size, void* d_ws, size_t ws_size,
                              hipStream_t stream) {
  (void)in_sizes; (void)n_in; (void)out_size; (void)ws_size;
  // Model constants
  const int NW = 1024, TW = 64;   // word level: B*S sequences, T steps
  const int NS = 32,  TS = 32;    // sentence level
  const int MW = NW * TW;         // 65536 attention rows (word)
  const int MS = NS * TS;         // 1024 attention rows (sentence)

  // ---- inputs (setup_inputs order) ----
  const int*   x     = (const int*)d_in[0];
  const int*   qids  = (const int*)d_in[1];
  const float* emb   = (const float*)d_in[2];
  const float* wih_f = (const float*)d_in[3];
  const float* whh_f = (const float*)d_in[4];
  const float* bih_f = (const float*)d_in[5];
  const float* bhh_f = (const float*)d_in[6];
  const float* wih_b = (const float*)d_in[7];
  const float* whh_b = (const float*)d_in[8];
  const float* bih_b = (const float*)d_in[9];
  const float* bhh_b = (const float*)d_in[10];
  const float* W1w   = (const float*)d_in[11];
  const float* b1w   = (const float*)d_in[12];
  const float* w2w   = (const float*)d_in[13];
  const float* Wqw   = (const float*)d_in[14];
  const float* bqw   = (const float*)d_in[15];
  const float* sih_f = (const float*)d_in[16];
  const float* shh_f = (const float*)d_in[17];
  const float* sbih_f= (const float*)d_in[18];
  const float* sbhh_f= (const float*)d_in[19];
  const float* sih_b = (const float*)d_in[20];
  const float* shh_b = (const float*)d_in[21];
  const float* sbih_b= (const float*)d_in[22];
  const float* sbhh_b= (const float*)d_in[23];
  const float* W1s   = (const float*)d_in[24];
  const float* b1s   = (const float*)d_in[25];
  const float* w2s   = (const float*)d_in[26];
  const float* b2s   = (const float*)d_in[27];
  const float* Wqs   = (const float*)d_in[28];
  const float* bqs   = (const float*)d_in[29];
  const float* Wfc   = (const float*)d_in[30];
  const float* bfc   = (const float*)d_in[31];

  // ---- workspace carve-out (256-B aligned) ----
  size_t off = 0;
  auto carve = [&](size_t bytes) -> void* {
    void* p = (char*)d_ws + off;
    off = (off + bytes + 255) & ~(size_t)255;
    return p;
  };
  _Float16* xe16   = (_Float16*)carve((size_t)MW * 256 * 2);   // [1024,64,256]
  _Float16* f16b   = (_Float16*)carve((size_t)MW * 256 * 2);   // word bi-GRU out
  float*    H      = (float*)   carve((size_t)NW * 128 * 4);
  _Float16* H16    = (_Float16*)carve((size_t)NW * 128 * 2);
  float*    qe     = (float*)   carve(5 * 256 * 4);
  float*    stw    = (float*)   carve((size_t)MW * 4);
  float*    sqw    = (float*)   carve((size_t)5 * MW * 4);
  float*    amw    = (float*)   carve((size_t)MW * 4);
  float*    sv     = (float*)   carve((size_t)NW * 256 * 4);   // sent_vec f32
  _Float16* sv16   = (_Float16*)carve((size_t)NW * 256 * 2);
  _Float16* g16    = (_Float16*)carve((size_t)MS * 256 * 2);   // sentence bi-GRU out
  float*    sts    = (float*)   carve((size_t)MS * 4);
  float*    sqs    = (float*)   carve((size_t)5 * MS * 4);
  float*    ams    = (float*)   carve((size_t)MS * 4);
  float*    aqs    = (float*)   carve((size_t)5 * MS * 4);     // softmaxed a_q
  float*    doc    = (float*)   carve((size_t)NS * 256 * 4);
  _Float16* wihf16 = (_Float16*)carve(384 * 256 * 2);
  _Float16* whhf16 = (_Float16*)carve(384 * 128 * 2);
  _Float16* wihb16 = (_Float16*)carve(384 * 256 * 2);
  _Float16* whhb16 = (_Float16*)carve(384 * 128 * 2);
  _Float16* sihf16 = (_Float16*)carve(384 * 256 * 2);
  _Float16* shhf16 = (_Float16*)carve(384 * 128 * 2);
  _Float16* sihb16 = (_Float16*)carve(384 * 256 * 2);
  _Float16* shhb16 = (_Float16*)carve(384 * 128 * 2);
  _Float16* W1w16  = (_Float16*)carve(256 * 256 * 2);
  _Float16* Wqw16  = (_Float16*)carve((size_t)5 * 256 * 256 * 2);
  _Float16* W1s16  = (_Float16*)carve(256 * 256 * 2);
  _Float16* Wqs16  = (_Float16*)carve((size_t)5 * 256 * 256 * 2);

  auto cv = [&](const float* s, _Float16* d, int n) {
    k_f32_to_f16<<<(n + 255) / 256, 256, 0, stream>>>(s, d, n);
  };

  // ---- f32 -> f16 weight packs ----
  cv(wih_f, wihf16, 384 * 256); cv(whh_f, whhf16, 384 * 128);
  cv(wih_b, wihb16, 384 * 256); cv(whh_b, whhb16, 384 * 128);
  cv(sih_f, sihf16, 384 * 256); cv(shh_f, shhf16, 384 * 128);
  cv(sih_b, sihb16, 384 * 256); cv(shh_b, shhb16, 384 * 128);
  cv(W1w, W1w16, 256 * 256); cv(Wqw, Wqw16, 5 * 256 * 256);
  cv(W1s, W1s16, 256 * 256); cv(Wqs, Wqs16, 5 * 256 * 256);

  // ---- embeddings ----
  k_embed_gather<<<MW, 256, 0, stream>>>(x, emb, xe16);
  k_query_emb<<<5, 256, 0, stream>>>(qids, emb, qe);

  // ---- word-level bi-GRU ----
  hipMemsetAsync(H, 0, (size_t)NW * 128 * 4, stream);
  hipMemsetAsync(H16, 0, (size_t)NW * 128 * 2, stream);
  for (int t = 0; t < TW; ++t)
    k_gru_step<<<NW / 16, 256, 0, stream>>>(xe16, TW, t, wihf16, whhf16, bih_f, bhh_f,
                                            H, H16, f16b, 0);
  hipMemsetAsync(H, 0, (size_t)NW * 128 * 4, stream);
  hipMemsetAsync(H16, 0, (size_t)NW * 128 * 2, stream);
  for (int i = 0; i < TW; ++i)
    k_gru_step<<<NW / 16, 256, 0, stream>>>(xe16, TW, TW - 1 - i, wihb16, whhb16, bih_b, bhh_b,
                                            H, H16, f16b, 128);

  // ---- word-level attention ----
  k_attn_score<<<MW / 16, 32, 0, stream>>>(f16b, W1w16, b1w, w2w, nullptr, stw);
  for (int q = 0; q < 5; ++q)
    k_attn_score<<<MW / 16, 32, 0, stream>>>(f16b, Wqw16 + (size_t)q * 256 * 256,
                                             bqw + q * 256, qe + q * 256, nullptr,
                                             sqw + (size_t)q * MW);
  k_softmax_mean<<<NW, TW, 0, stream>>>(stw, sqw, amw, nullptr, NW, TW);
  k_pool<<<NW, 256, 0, stream>>>(amw, f16b, sv, sv16, TW);

  // ---- sentence-level bi-GRU ----
  hipMemsetAsync(H, 0, (size_t)NS * 128 * 4, stream);
  hipMemsetAsync(H16, 0, (size_t)NS * 128 * 2, stream);
  for (int s = 0; s < TS; ++s)
    k_gru_step<<<NS / 16, 256, 0, stream>>>(sv16, TS, s, sihf16, shhf16, sbih_f, sbhh_f,
                                            H, H16, g16, 0);
  hipMemsetAsync(H, 0, (size_t)NS * 128 * 4, stream);
  hipMemsetAsync(H16, 0, (size_t)NS * 128 * 2, stream);
  for (int i = 0; i < TS; ++i)
    k_gru_step<<<NS / 16, 256, 0, stream>>>(sv16, TS, TS - 1 - i, sihb16, shhb16, sbih_b, sbhh_b,
                                            H, H16, g16, 128);

  // ---- sentence-level attention ----
  k_attn_score<<<MS / 16, 32, 0, stream>>>(g16, W1s16, b1s, w2s, b2s, sts);
  for (int q = 0; q < 5; ++q)
    k_attn_score<<<MS / 16, 32, 0, stream>>>(g16, Wqs16 + (size_t)q * 256 * 256,
                                             bqs + q * 256, qe + q * 256, nullptr,
                                             sqs + (size_t)q * MS);
  k_softmax_mean<<<NS, TS, 0, stream>>>(sts, sqs, ams, aqs, NS, TS);
  k_pool<<<NS, 256, 0, stream>>>(ams, g16, doc, nullptr, TS);

  // ---- classifier + all_q_value ----
  k_head<<<1, 64, 0, stream>>>(doc, Wfc, bfc, aqs, (float*)d_out);
}
